// AttentionBlock_17428977287176
// MI455X (gfx1250) — matrix operand order repaired and checked by reading, not compiled, not run on hardware
//
#include <hip/hip_runtime.h>
#include <hip/hip_fp16.h>
#include <stdint.h>

typedef __attribute__((ext_vector_type(16))) _Float16 v16h;
typedef __attribute__((ext_vector_type(8)))  float    v8f;
typedef __attribute__((ext_vector_type(2)))  _Float16 h2;

#define C_DIM  512
#define N_POS  4096            // 64*64
#define BATCH  4
#define NGRP   32
#define CG     16
#define ROWS_TOTAL (BATCH * N_POS)   // 16384

// padded LDS strides (halves) so 16 consecutive rows map to distinct banks
#define QS_STRIDE 520          // 260 words -> step 4 banks
#define PS_STRIDE 136          // 68 words  -> step 4 banks
#define PANEL_STRIDE 72        // 36 words  -> step 36 banks

// ---- WMMA fragment index helpers (ISA 7.12.2, 16-bit A 16x32 / B 32x16) ----
__device__ __forceinline__ int a_kbase(int v, int half) {
    return ((v >> 2) << 4) + (half << 3) + ((v & 3) << 1);
}
__device__ __forceinline__ int b_kbase(int v, int kgrp) {
    return ((v >> 2) << 4) + (kgrp << 3) + ((v & 3) << 1);
}

// ---- CDNA5 async global->LDS copy (per-lane 16B, tracked by ASYNCcnt) ----
__device__ __forceinline__ void async_copy_b128(uint32_t lds_addr, const void* gaddr) {
    asm volatile("global_load_async_to_lds_b128 %0, %1, off"
                 :: "v"(lds_addr), "v"(gaddr) : "memory");
}
#define WAIT_ASYNC(imm) asm volatile("s_wait_asynccnt " #imm ::: "memory")

__device__ __forceinline__ uint32_t lds_off(const void* p) {
    // generic pointers to LDS carry the byte offset in the low 32 bits
    return (uint32_t)(uintptr_t)p;
}

// =====================  GroupNorm  =====================
__global__ __launch_bounds__(256) void gn_stats_kernel(const float* __restrict__ x,
                                                       float* __restrict__ stats) {
    __shared__ float ssum[256], ssq[256];
    const int bg = blockIdx.x;
    const int b  = bg >> 5;
    const int g  = bg & 31;
    const size_t base = (size_t)b * N_POS * C_DIM + (size_t)g * CG;
    float s = 0.f, s2 = 0.f;
    for (int i = threadIdx.x; i < N_POS * CG; i += 256) {
        const int hw = i >> 4;
        const int c  = i & 15;
        const float v = x[base + (size_t)hw * C_DIM + c];
        s += v; s2 += v * v;
    }
    ssum[threadIdx.x] = s; ssq[threadIdx.x] = s2;
    __syncthreads();
    for (int off = 128; off > 0; off >>= 1) {
        if (threadIdx.x < off) {
            ssum[threadIdx.x] += ssum[threadIdx.x + off];
            ssq [threadIdx.x] += ssq [threadIdx.x + off];
        }
        __syncthreads();
    }
    if (threadIdx.x == 0) {
        const float inv = 1.0f / (float)(N_POS * CG);
        const float mu  = ssum[0] * inv;
        const float var = ssq[0] * inv - mu * mu;
        stats[2 * bg]     = mu;
        stats[2 * bg + 1] = rsqrtf(var + 1e-5f);
    }
}

__global__ __launch_bounds__(256) void gn_apply_kernel(const float* __restrict__ x,
                                                       const float* __restrict__ gamma,
                                                       const float* __restrict__ beta,
                                                       const float* __restrict__ stats,
                                                       _Float16* __restrict__ xn) {
    const int idx = blockIdx.x * 256 + threadIdx.x;
    const int c   = idx & (C_DIM - 1);
    const int row = idx >> 9;
    const int b   = row >> 12;
    const int bg  = b * NGRP + (c >> 4);
    const float mu   = stats[2 * bg];
    const float rstd = stats[2 * bg + 1];
    xn[idx] = (_Float16)((x[idx] - mu) * rstd * gamma[c] + beta[c]);
}

// =====================  weight fp32 -> f16 transposed  =====================
__global__ __launch_bounds__(256) void wconv_kernel(const float* __restrict__ w,
                                                    _Float16* __restrict__ wT) {
    const int idx = blockIdx.x * 256 + threadIdx.x;
    const int k = idx >> 9, n = idx & (C_DIM - 1);
    wT[n * C_DIM + k] = (_Float16)w[k * C_DIM + n];
}

// =====================  128x64-tile GEMM with async-LDS double buffering ====
// C[16384,512-slice] = A[16384,512] @ W + bias ; epilogue by mode:
//   0: f16 row-major * scale (Q)   1: f16 row-major (K)
//   2: f16 transposed per batch (V)   3: fp32 + bias + residual (proj)
__global__ __launch_bounds__(256) void gemm128_kernel(const _Float16* __restrict__ A,
                                                      const _Float16* __restrict__ wT,
                                                      const float* __restrict__ bias,
                                                      _Float16* __restrict__ out_h,
                                                      float* __restrict__ out_f,
                                                      const float* __restrict__ resid,
                                                      int mode, float scale) {
    __shared__ _Float16 As[2][128 * PANEL_STRIDE];   // 2 x 18 KB
    __shared__ _Float16 Bs[2][64 * PANEL_STRIDE];    // 2 x 9 KB

    const int tid  = threadIdx.x;
    const int lane = tid & 31;
    const int wave = tid >> 5;
    const int half = lane >> 4;
    const int n    = lane & 15;
    const int m0   = blockIdx.x * 128;
    const int n0   = blockIdx.y * 64;

    auto issue_panel = [&](int kt, int buf) {
        const _Float16* ga = A + (size_t)m0 * C_DIM + kt * 64;
#pragma unroll
        for (int i = 0; i < 4; ++i) {
            const int idx = i * 256 + tid;
            const int row = idx >> 3, seg = idx & 7;
            async_copy_b128(lds_off(&As[buf][row * PANEL_STRIDE + seg * 8]),
                            ga + (size_t)row * C_DIM + seg * 8);
        }
        const _Float16* gb = wT + (size_t)n0 * C_DIM + kt * 64;
#pragma unroll
        for (int i = 0; i < 2; ++i) {
            const int idx = i * 256 + tid;
            const int col = idx >> 3, seg = idx & 7;
            async_copy_b128(lds_off(&Bs[buf][col * PANEL_STRIDE + seg * 8]),
                            gb + (size_t)col * C_DIM + seg * 8);
        }
    };

    v8f acc[4];
    const v8f zero = {};
#pragma unroll
    for (int s = 0; s < 4; ++s) acc[s] = zero;

    issue_panel(0, 0);
    for (int kt = 0; kt < 8; ++kt) {
        const int buf = kt & 1;
        if (kt < 7) {
            issue_panel(kt + 1, buf ^ 1);
            WAIT_ASYNC(0x6);          // panel kt done; kt+1 (6 ops) still in flight
        } else {
            WAIT_ASYNC(0x0);
        }
        __syncthreads();
#pragma unroll
        for (int kk = 0; kk < 64; kk += 32) {
            v16h a;
#pragma unroll
            for (int v = 0; v < 8; ++v) {
                h2 pa = *(const h2*)(&As[buf][(wave * 16 + n) * PANEL_STRIDE + kk + a_kbase(v, half)]);
                a[2 * v] = pa.x; a[2 * v + 1] = pa.y;
            }
#pragma unroll
            for (int s = 0; s < 4; ++s) {
                v16h bb;
#pragma unroll
                for (int v = 0; v < 8; ++v) {
                    h2 pb = *(const h2*)(&Bs[buf][(s * 16 + n) * PANEL_STRIDE + kk + b_kbase(v, half)]);
                    bb[2 * v] = pb.x; bb[2 * v + 1] = pb.y;
                }
                acc[s] = __builtin_amdgcn_wmma_f32_16x16x32_f16(false, a, false, bb,
                                                                (short)0, acc[s], false, false);
            }
        }
        __syncthreads();              // panel buf free before it is re-filled
    }

#pragma unroll
    for (int s = 0; s < 4; ++s) {
        const int col = n0 + s * 16 + n;
        const float bv = bias[col];
#pragma unroll
        for (int r = 0; r < 8; ++r) {
            const int row = m0 + wave * 16 + r + 8 * half;
            if (mode == 3) {
                const size_t idx = (size_t)row * C_DIM + col;
                out_f[idx] = acc[s][r] + bv + resid[idx];
            } else {
                const float val = (acc[s][r] + bv) * scale;
                if (mode == 2) {
                    const int b = row >> 12, pos = row & (N_POS - 1);
                    out_h[((size_t)(b * C_DIM + col)) * N_POS + pos] = (_Float16)val;
                } else {
                    out_h[(size_t)row * C_DIM + col] = (_Float16)val;
                }
            }
        }
    }
}

// =====================  Flash attention (WMMA)  =====================
__global__ __launch_bounds__(256) void attn_kernel(const _Float16* __restrict__ q_h,
                                                   const _Float16* __restrict__ k_h,
                                                   const _Float16* __restrict__ vT_h,
                                                   _Float16* __restrict__ ao_h) {
    __shared__ _Float16 Qs[16 * QS_STRIDE];   // padded: conflict-free A-frag reads
    __shared__ _Float16 Ps[16 * PS_STRIDE];
    __shared__ float m_state[16], l_state[16], scale_s[16], newm_s[16];
    __shared__ float wmax[8][16], wsum[8][16];

    const int tid  = threadIdx.x;
    const int lane = tid & 31;
    const int wave = tid >> 5;
    const int half = lane >> 4;
    const int n    = lane & 15;
    const int b    = blockIdx.y;
    const int row0 = b * N_POS + blockIdx.x * 16;

    // stage Q tile (pre-scaled by C^-0.5) into padded LDS via async copies
    {
        const _Float16* src = q_h + (size_t)row0 * C_DIM;
#pragma unroll
        for (int i = 0; i < 4; ++i) {
            const int idx = i * 256 + tid;
            const int row = idx >> 6, seg = idx & 63;
            async_copy_b128(lds_off(&Qs[row * QS_STRIDE + seg * 8]),
                            src + (size_t)row * C_DIM + seg * 8);
        }
        WAIT_ASYNC(0x0);
    }
    if (tid < 16) { m_state[tid] = -1e30f; l_state[tid] = 0.0f; }
    __syncthreads();

    v8f acc[4];
    const v8f zero = {};
#pragma unroll
    for (int s = 0; s < 4; ++s) acc[s] = zero;

    for (int j0 = 0; j0 < N_POS; j0 += 128) {
        // ---- Phase A: S(16x16) = Q x K^T, K-fragments double-buffered ----
        v8f S = zero;
        const _Float16* kb = k_h + (size_t)(b * N_POS + j0 + wave * 16 + n) * C_DIM;
        v16h bb0, bb1;
#pragma unroll
        for (int v = 0; v < 8; ++v) {
            h2 pb = *(const h2*)(kb + b_kbase(v, half));
            bb0[2 * v] = pb.x; bb0[2 * v + 1] = pb.y;
        }
#pragma unroll
        for (int kk = 0; kk < C_DIM; kk += 32) {
            if (kk + 32 < C_DIM) {
#pragma unroll
                for (int v = 0; v < 8; ++v) {
                    h2 pb = *(const h2*)(kb + kk + 32 + b_kbase(v, half));
                    bb1[2 * v] = pb.x; bb1[2 * v + 1] = pb.y;
                }
            }
            v16h a;
#pragma unroll
            for (int v = 0; v < 8; ++v) {
                h2 pa = *(const h2*)(&Qs[n * QS_STRIDE + kk + a_kbase(v, half)]);
                a[2 * v] = pa.x; a[2 * v + 1] = pa.y;
            }
            S = __builtin_amdgcn_wmma_f32_16x16x32_f16(false, a, false, bb0,
                                                       (short)0, S, false, false);
            bb0 = bb1;
        }
        // per-wave row max across the 16 N-lanes of each half-group
#pragma unroll
        for (int r = 0; r < 8; ++r) {
            float mr = S[r];
            mr = fmaxf(mr, __shfl_xor(mr, 1));
            mr = fmaxf(mr, __shfl_xor(mr, 2));
            mr = fmaxf(mr, __shfl_xor(mr, 4));
            mr = fmaxf(mr, __shfl_xor(mr, 8));
            if (n == 0) wmax[wave][r + 8 * half] = mr;
        }
        __syncthreads();
        if (tid < 16) {
            float mx = m_state[tid];
#pragma unroll
            for (int w = 0; w < 8; ++w) mx = fmaxf(mx, wmax[w][tid]);
            newm_s[tid]  = mx;
            scale_s[tid] = __expf(m_state[tid] - mx);
            m_state[tid] = mx;
        }
        __syncthreads();
        // ---- P = exp(S - m_new): f16 P tile + partial row sums ----
#pragma unroll
        for (int r = 0; r < 8; ++r) {
            const int row = r + 8 * half;
            const float p = __expf(S[r] - newm_s[row]);
            Ps[row * PS_STRIDE + wave * 16 + n] = (_Float16)p;
            float sr = p;
            sr += __shfl_xor(sr, 1);
            sr += __shfl_xor(sr, 2);
            sr += __shfl_xor(sr, 4);
            sr += __shfl_xor(sr, 8);
            if (n == 0) wsum[wave][row] = sr;
        }
        __syncthreads();
        if (tid < 16) {
            float l = l_state[tid] * scale_s[tid];
#pragma unroll
            for (int w = 0; w < 8; ++w) l += wsum[w][tid];
            l_state[tid] = l;
        }
        // prefetch next K block (L2-resident; global_prefetch_b8)
        if (j0 + 128 < N_POS) {
            const char* nk = (const char*)(k_h +
                (size_t)(b * N_POS + j0 + 128 + wave * 16 + n) * C_DIM);
            __builtin_prefetch(nk + half * 512, 0, 0);
            __builtin_prefetch(nk + half * 512 + 256, 0, 0);
        }
        // ---- Phase B: rescale acc, acc += P(16x128) x V-slice, V prefetched ----
#pragma unroll
        for (int s = 0; s < 4; ++s)
#pragma unroll
            for (int r = 0; r < 8; ++r)
                acc[s][r] *= scale_s[r + 8 * half];

        const size_t vbase = (size_t)(b * C_DIM) * N_POS + j0;
#pragma unroll
        for (int kk = 0; kk < 128; kk += 32) {
            v16h a;
#pragma unroll
            for (int v = 0; v < 8; ++v) {
                h2 pa = *(const h2*)(&Ps[n * PS_STRIDE + kk + a_kbase(v, half)]);
                a[2 * v] = pa.x; a[2 * v + 1] = pa.y;
            }
            v16h bb;
            {
                const _Float16* vp = vT_h + vbase + (size_t)(wave * 64 + n) * N_POS + kk;
#pragma unroll
                for (int v = 0; v < 8; ++v) {
                    h2 pb = *(const h2*)(vp + b_kbase(v, half));
                    bb[2 * v] = pb.x; bb[2 * v + 1] = pb.y;
                }
            }
#pragma unroll
            for (int s = 0; s < 4; ++s) {
                v16h bbn;
                if (s < 3) {
                    const _Float16* vp = vT_h + vbase +
                        (size_t)(wave * 64 + (s + 1) * 16 + n) * N_POS + kk;
#pragma unroll
                    for (int v = 0; v < 8; ++v) {
                        h2 pb = *(const h2*)(vp + b_kbase(v, half));
                        bbn[2 * v] = pb.x; bbn[2 * v + 1] = pb.y;
                    }
                }
                acc[s] = __builtin_amdgcn_wmma_f32_16x16x32_f16(false, a, false, bb,
                                                                (short)0, acc[s], false, false);
                if (s < 3) bb = bbn;
            }
        }
    }
    __syncthreads();
#pragma unroll
    for (int s = 0; s < 4; ++s) {
        const int chan = wave * 64 + s * 16 + n;
#pragma unroll
        for (int r = 0; r < 8; ++r) {
            const int row = r + 8 * half;
            const float val = acc[s][r] / l_state[row];
            ao_h[(size_t)(row0 + row) * C_DIM + chan] = (_Float16)val;
        }
    }
}

// =====================  host launcher  =====================
extern "C" void kernel_launch(void* const* d_in, const int* in_sizes, int n_in,
                              void* d_out, int out_size, void* d_ws, size_t ws_size,
                              hipStream_t stream) {
    const float* x     = (const float*)d_in[0];
    const float* gamma = (const float*)d_in[1];
    const float* beta  = (const float*)d_in[2];
    const float* wq    = (const float*)d_in[3];
    const float* bq    = (const float*)d_in[4];
    const float* wk    = (const float*)d_in[5];
    const float* bk    = (const float*)d_in[6];
    const float* wv    = (const float*)d_in[7];
    const float* bv    = (const float*)d_in[8];
    const float* wp    = (const float*)d_in[9];
    const float* bp    = (const float*)d_in[10];
    float* out = (float*)d_out;

    char* ws = (char*)d_ws;
    const size_t SZ_ACT = (size_t)ROWS_TOTAL * C_DIM * sizeof(_Float16);  // 16 MB
    const size_t SZ_W   = (size_t)C_DIM * C_DIM * sizeof(_Float16);       // 512 KB
    size_t off = 0;
    float*    stats = (float*)(ws + off);    off += 4096;
    _Float16* xn_h  = (_Float16*)(ws + off); off += SZ_ACT;
    _Float16* q_h   = (_Float16*)(ws + off); off += SZ_ACT;
    _Float16* k_h   = (_Float16*)(ws + off); off += SZ_ACT;
    _Float16* vT_h  = (_Float16*)(ws + off); off += SZ_ACT;
    _Float16* ao_h  = (_Float16*)(ws + off); off += SZ_ACT;
    _Float16* wqT   = (_Float16*)(ws + off); off += SZ_W;
    _Float16* wkT   = (_Float16*)(ws + off); off += SZ_W;
    _Float16* wvT   = (_Float16*)(ws + off); off += SZ_W;
    _Float16* wpT   = (_Float16*)(ws + off); off += SZ_W;

    gn_stats_kernel<<<BATCH * NGRP, 256, 0, stream>>>(x, stats);
    gn_apply_kernel<<<(ROWS_TOTAL * C_DIM) / 256, 256, 0, stream>>>(x, gamma, beta, stats, xn_h);

    wconv_kernel<<<(C_DIM * C_DIM) / 256, 256, 0, stream>>>(wq, wqT);
    wconv_kernel<<<(C_DIM * C_DIM) / 256, 256, 0, stream>>>(wk, wkT);
    wconv_kernel<<<(C_DIM * C_DIM) / 256, 256, 0, stream>>>(wv, wvT);
    wconv_kernel<<<(C_DIM * C_DIM) / 256, 256, 0, stream>>>(wp, wpT);

    const dim3 ggrid(ROWS_TOTAL / 128, C_DIM / 64);   // 128 x 8
    const float qscale = 0.04419417382415922f;        // 512^-0.5
    gemm128_kernel<<<ggrid, 256, 0, stream>>>(xn_h, wqT, bq, q_h, nullptr, nullptr, 0, qscale);
    gemm128_kernel<<<ggrid, 256, 0, stream>>>(xn_h, wkT, bk, k_h, nullptr, nullptr, 1, 1.0f);
    gemm128_kernel<<<ggrid, 256, 0, stream>>>(xn_h, wvT, bv, vT_h, nullptr, nullptr, 2, 1.0f);

    attn_kernel<<<dim3(N_POS / 16, BATCH), 256, 0, stream>>>(q_h, k_h, vT_h, ao_h);

    gemm128_kernel<<<ggrid, 256, 0, stream>>>(ao_h, wpT, bp, nullptr, out, x, 3, 1.0f);
}
